// FusionAttentionModel_3736621547774
// MI455X (gfx1250) — compile-verified, matrix operand order checked by reading, not run on hardware
//
#include <hip/hip_runtime.h>

#define DD   256
#define NB   16384
#define NKB  136          // 4352 / 32 k-blocks
#define NC   26
#define MBLK 64           // rows per workgroup in main GEMM
#define NBLK (NB / MBLK)  // 256 workgroups

typedef __attribute__((ext_vector_type(16))) _Float16 v16h;
typedef __attribute__((ext_vector_type(8)))  _Float16 v8h;
typedef __attribute__((ext_vector_type(8)))  float    v8f;
typedef int v4i __attribute__((vector_size(16)));

#if defined(__has_builtin)
#if __has_builtin(__builtin_amdgcn_global_load_async_to_lds_b128) && __has_builtin(__builtin_amdgcn_s_wait_asynccnt)
#define USE_ASYNC_LDS 1
#endif
#endif

// ---------------------------------------------------------------------------
// helpers
// ---------------------------------------------------------------------------
__device__ inline v8h cvt8(float4 a, float4 b) {
    v8h r;
    r[0] = (_Float16)a.x; r[1] = (_Float16)a.y; r[2] = (_Float16)a.z; r[3] = (_Float16)a.w;
    r[4] = (_Float16)b.x; r[5] = (_Float16)b.y; r[6] = (_Float16)b.z; r[7] = (_Float16)b.w;
    return r;
}

// 16-bit 16x32 fragment per CDNA5 WMMA layout:
// lanes 0-15 (row = lane): k[0..8) and k[16..24); lanes 16-31 (row = lane-16): k[8..16) and k[24..32)
__device__ inline v16h load_frag(const _Float16* base, int lane) {
    const int m  = lane & 15;
    const int hi = lane >> 4;
    const _Float16* p = base + m * 32 + hi * 8;
    v8h lo = *(const v8h*)p;
    v8h hh = *(const v8h*)(p + 16);
    return __builtin_shufflevector(lo, hh, 0,1,2,3,4,5,6,7,8,9,10,11,12,13,14,15);
}

__device__ inline const float* seg_select(int kg,
                                          const float* xc, const float* xb,
                                          const float* xf, const float* xt,
                                          int& width, int& koff) {
    if      (kg < 2048) { width = 2048; koff = kg;        return xc; }
    else if (kg < 3072) { width = 1024; koff = kg - 2048; return xb; }
    else if (kg < 3584) { width = 512;  koff = kg - 3072; return xf; }
    else                { width = 768;  koff = kg - 3584; return xt; }
}

// stage A: 64 rows x 32 k, f32 -> f16.  4 threads per row, 8 floats each.
__device__ inline void stage_A(const float* src, int width, int koff,
                               int rowbase, int row, int q, _Float16* bufA) {
    const float4* g = (const float4*)(src + (size_t)(rowbase + row) * width + koff + q * 8);
    float4 f0 = g[0], f1 = g[1];
    *(v8h*)&bufA[row * 32 + q * 8] = cvt8(f0, f1);
}

// stage B: 256 cols x 32 k f16 block (16 KB contiguous) -> LDS, async DMA
__device__ inline void stage_B(const _Float16* wbig, int kb, int t, _Float16* bufB) {
    _Float16* gB = const_cast<_Float16*>(wbig) + (size_t)kb * (DD * 32) + t * 32;
    _Float16* lB = &bufB[t * 32];
#ifdef USE_ASYNC_LDS
#pragma unroll
    for (int q = 0; q < 4; ++q)
        __builtin_amdgcn_global_load_async_to_lds_b128(
            (__attribute__((address_space(1))) v4i*)(gB + q * 8),
            (__attribute__((address_space(3))) v4i*)(lB + q * 8), 0, 0);
#else
    const uint4* g4 = (const uint4*)gB;
    uint4*       l4 = (uint4*)lB;
    l4[0] = g4[0]; l4[1] = g4[1]; l4[2] = g4[2]; l4[3] = g4[3];
#endif
}

// ---------------------------------------------------------------------------
// prep kernels:  M = I + out_w @ w_v ;  P = fc1_w @ M ;  Wbig = 0.25 * P @ Wcat
// ---------------------------------------------------------------------------
__global__ __launch_bounds__(256) void k_prep_M(const float* __restrict__ out_w,
                                                const float* __restrict__ in_proj_w,
                                                float* __restrict__ M) {
    const int e = blockIdx.x * 256 + threadIdx.x;
    const int i = e >> 8, j = e & 255;
    const float* wv = in_proj_w + 2 * DD * DD;   // w_v = in_proj_w[2D:]
    float s = (i == j) ? 1.0f : 0.0f;
    for (int k = 0; k < DD; ++k) s = fmaf(out_w[i * DD + k], wv[k * DD + j], s);
    M[e] = s;
}

__global__ __launch_bounds__(256) void k_prep_P(const float* __restrict__ fc1_w,
                                                const float* __restrict__ M,
                                                float* __restrict__ P) {
    const int e = blockIdx.x * 256 + threadIdx.x;
    const int i = e >> 8, j = e & 255;
    float s = 0.0f;
    for (int k = 0; k < DD; ++k) s = fmaf(fc1_w[i * DD + k], M[k * DD + j], s);
    P[e] = s;
}

// Wbig stored k-blocked: wbig[((k>>5)*256 + j)*32 + (k&31)]  (f16)
__global__ __launch_bounds__(256) void k_prep_Wbig(const float* __restrict__ P,
                                                   const float* __restrict__ wc, const float* __restrict__ wb,
                                                   const float* __restrict__ wf, const float* __restrict__ wt,
                                                   _Float16* __restrict__ wbig) {
    const int j = blockIdx.x / 17;
    const int k = (blockIdx.x % 17) * 256 + threadIdx.x;
    int width, koff;
    const float* src = seg_select(k, wc, wb, wf, wt, width, koff);
    float s = 0.0f;
    for (int t = 0; t < DD; ++t) s = fmaf(P[j * DD + t], src[t * width + koff], s);
    wbig[((size_t)(k >> 5) * DD + j) * 32 + (k & 31)] = (_Float16)(0.25f * s);
}

// hb = P @ bbar + fc1_w @ (out_w @ b_v + out_b) + fc1_b
__global__ __launch_bounds__(256) void k_prep_bias(const float* __restrict__ out_w, const float* __restrict__ out_b,
                                                   const float* __restrict__ in_proj_b,
                                                   const float* __restrict__ fc1_w, const float* __restrict__ fc1_b,
                                                   const float* __restrict__ P,
                                                   const float* __restrict__ b_ctx, const float* __restrict__ b_body,
                                                   const float* __restrict__ b_face, const float* __restrict__ b_text,
                                                   float* __restrict__ hb) {
    __shared__ float c1[DD], bb[DD];
    const int i = threadIdx.x;
    const float* bv = in_proj_b + 2 * DD;
    float s = out_b[i];
    for (int k = 0; k < DD; ++k) s = fmaf(out_w[i * DD + k], bv[k], s);
    c1[i] = s;
    bb[i] = 0.25f * (b_ctx[i] + b_body[i] + b_face[i] + b_text[i]);
    __syncthreads();
    float h = fc1_b[i];
    for (int k = 0; k < DD; ++k) h = fmaf(fc1_w[i * DD + k], c1[k], h);
    for (int k = 0; k < DD; ++k) h = fmaf(P[i * DD + k], bb[k], h);
    hb[i] = h;
}

// ---------------------------------------------------------------------------
// main GEMM:  h_pre(16384x256) = Xcat(16384x4352) @ Wbig.T + hb    (f16 WMMA)
// double-buffered LDS; async DMA for B one k-step ahead of compute
// ---------------------------------------------------------------------------
__global__ __launch_bounds__(256) void k_main(const float* __restrict__ xc, const float* __restrict__ xb,
                                              const float* __restrict__ xf, const float* __restrict__ xt,
                                              const _Float16* __restrict__ wbig,
                                              const float* __restrict__ hb,
                                              float* __restrict__ hpre,
                                              float* __restrict__ psum, float* __restrict__ psq) {
    __shared__ _Float16 ldsA[2][MBLK * 32];    // 2 x 4 KB
    __shared__ _Float16 ldsB[2][DD * 32];      // 2 x 16 KB
    __shared__ float    s_psum[8 * DD];        // 8 KB
    __shared__ float    s_psq [8 * DD];        // 8 KB
    __shared__ float    s_hb[DD];              // 1 KB

    const int t       = threadIdx.x;
    const int wave    = t >> 5;
    const int lane    = t & 31;
    const int rowbase = blockIdx.x * MBLK;

    const int rgrp  = wave & 3;    // row group: rows rgrp*16 .. +15
    const int chalf = wave >> 2;   // column half: cols chalf*128 .. +127

    s_hb[t] = hb[t];

    v8f acc[8];
    v8f zz = {0.f,0.f,0.f,0.f,0.f,0.f,0.f,0.f};
#pragma unroll
    for (int jt = 0; jt < 8; ++jt) acc[jt] = zz;

    const int arow = t >> 2, aq = t & 3;   // 4 threads per X row, 8 f32 each

    // ---- prologue: stage k-block 0 into buffer 0
    {
        int width, koff;
        const float* src = seg_select(0, xc, xb, xf, xt, width, koff);
        stage_B(wbig, 0, t, ldsB[0]);
        stage_A(src, width, koff, rowbase, arow, aq, ldsA[0]);
    }
#ifdef USE_ASYNC_LDS
    __builtin_amdgcn_s_wait_asynccnt(0);
#endif
    __syncthreads();

#pragma unroll 1
    for (int kb = 0; kb < NKB; ++kb) {
        const int cur = kb & 1;

        // ---- stage next k-block into the other buffer (overlaps with compute)
        if (kb + 1 < NKB) {
            int width, koff;
            const float* src = seg_select((kb + 1) * 32, xc, xb, xf, xt, width, koff);
            stage_B(wbig, kb + 1, t, ldsB[cur ^ 1]);
            stage_A(src, width, koff, rowbase, arow, aq, ldsA[cur ^ 1]);
        }

        // ---- compute on current buffers: 16 rows x 128 cols per wave
        {
            v16h afrag = load_frag(&ldsA[cur][rgrp * 16 * 32], lane);
            const _Float16* Bb = &ldsB[cur][chalf * 128 * 32];
            v16h b0 = load_frag(Bb, lane);
#pragma unroll
            for (int jt = 0; jt < 7; ++jt) {
                v16h b1 = load_frag(&Bb[(jt + 1) * 16 * 32], lane);   // prefetch next tile
                acc[jt] = __builtin_amdgcn_wmma_f32_16x16x32_f16(
                    false, afrag, false, b0, (short)0, acc[jt], false, false);
                b0 = b1;
            }
            acc[7] = __builtin_amdgcn_wmma_f32_16x16x32_f16(
                false, afrag, false, b0, (short)0, acc[7], false, false);
        }

#ifdef USE_ASYNC_LDS
        if (kb + 1 < NKB) __builtin_amdgcn_s_wait_asynccnt(0);
#endif
        __syncthreads();
    }

    // ---- epilogue: add bias, store h_pre, deterministic per-block column sums
    const int hi = lane >> 4;
    const int n  = lane & 15;
#pragma unroll
    for (int jt = 0; jt < 8; ++jt) {
        const int col  = chalf * 128 + jt * 16 + n;
        const float bias = s_hb[col];
        float ssum = 0.f, ssq = 0.f;
        float* hp = hpre + (size_t)(rowbase + rgrp * 16 + hi * 8) * DD + col;
#pragma unroll
        for (int v = 0; v < 8; ++v) {
            float val = acc[jt][v] + bias;
            hp[(size_t)v * DD] = val;
            ssum += val;
            ssq  += val * val;
        }
        s_psum[(rgrp * 2 + hi) * DD + col] = ssum;   // unique owner per slot: no races
        s_psq [(rgrp * 2 + hi) * DD + col] = ssq;
    }
    __syncthreads();
    float a = 0.f, b = 0.f;
#pragma unroll
    for (int p = 0; p < 8; ++p) { a += s_psum[p * DD + t]; b += s_psq[p * DD + t]; }
    psum[blockIdx.x * DD + t] = a;
    psq [blockIdx.x * DD + t] = b;
}

// ---------------------------------------------------------------------------
// BN stats -> fused scale/shift
// ---------------------------------------------------------------------------
__global__ __launch_bounds__(256) void k_stats(const float* __restrict__ psum, const float* __restrict__ psq,
                                               const float* __restrict__ bn_g, const float* __restrict__ bn_b,
                                               float* __restrict__ scale, float* __restrict__ shift) {
    const int t = threadIdx.x;
    float s = 0.f, q = 0.f;
    for (int blk = 0; blk < NBLK; ++blk) { s += psum[blk * DD + t]; q += psq[blk * DD + t]; }
    const float mu  = s * (1.0f / NB);
    const float var = q * (1.0f / NB) - mu * mu;
    const float r   = rsqrtf(var + 1e-5f);
    const float sc  = bn_g[t] * r;
    scale[t] = sc;
    shift[t] = bn_b[t] - mu * sc;
}

// ---------------------------------------------------------------------------
// head: out = relu(BN(h_pre)) @ fc2.T + fc2_b
// ---------------------------------------------------------------------------
__global__ __launch_bounds__(256) void k_head(const float* __restrict__ hpre,
                                              const float* __restrict__ scale, const float* __restrict__ shift,
                                              const float* __restrict__ fc2_w, const float* __restrict__ fc2_b,
                                              float* __restrict__ out) {
    __shared__ float s_scale[DD], s_shift[DD], s_w[NC * DD];
    const int t = threadIdx.x;
    s_scale[t] = scale[t];
    s_shift[t] = shift[t];
    for (int q = t; q < NC * DD; q += 256) s_w[q] = fc2_w[q];
    __syncthreads();

    const int r = blockIdx.x * 8 + (t >> 5);
    const int c = t & 31;
    if (c < NC) {
        const float* hp = hpre + (size_t)r * DD;
        const float* w  = &s_w[c * DD];
        float s = fc2_b[c];
        for (int k = 0; k < DD; ++k) {
            float v = fmaf(hp[k], s_scale[k], s_shift[k]);
            v = v > 0.f ? v : 0.f;
            s = fmaf(v, w[k], s);
        }
        out[(size_t)r * NC + c] = s;
    }
}

// ---------------------------------------------------------------------------
extern "C" void kernel_launch(void* const* d_in, const int* in_sizes, int n_in,
                              void* d_out, int out_size, void* d_ws, size_t ws_size,
                              hipStream_t stream) {
    const float* x_ctx  = (const float*)d_in[0];
    const float* x_body = (const float*)d_in[1];
    const float* x_face = (const float*)d_in[2];
    const float* x_text = (const float*)d_in[3];
    const float* w_ctx  = (const float*)d_in[4];
    const float* b_ctx  = (const float*)d_in[5];
    const float* w_body = (const float*)d_in[6];
    const float* b_body = (const float*)d_in[7];
    const float* w_face = (const float*)d_in[8];
    const float* b_face = (const float*)d_in[9];
    const float* w_text = (const float*)d_in[10];
    const float* b_text = (const float*)d_in[11];
    const float* in_proj_w = (const float*)d_in[12];
    const float* in_proj_b = (const float*)d_in[13];
    const float* out_w  = (const float*)d_in[14];
    const float* out_b  = (const float*)d_in[15];
    // d_in[16..19] = gate params: mathematically dead (feats_cross == feats_self)
    const float* fc1_w  = (const float*)d_in[20];
    const float* fc1_b  = (const float*)d_in[21];
    const float* bn_g   = (const float*)d_in[22];
    const float* bn_b   = (const float*)d_in[23];
    const float* fc2_w  = (const float*)d_in[24];
    const float* fc2_b  = (const float*)d_in[25];
    float* outp = (float*)d_out;

    char* ws = (char*)d_ws;
    _Float16* wbig = (_Float16*)ws;                       // 256*4352*2 = 2,228,224 B
    float* Mbuf  = (float*)(ws + (size_t)DD * 4352 * 2);  // 256 KB
    float* Pbuf  = Mbuf + DD * DD;                        // 256 KB
    float* hb    = Pbuf + DD * DD;                        // 1 KB
    float* psum  = hb + DD;                               // 256*256*4 = 256 KB
    float* psq   = psum + NBLK * DD;                      // 256 KB
    float* scale = psq + NBLK * DD;
    float* shift = scale + DD;
    float* hpre  = shift + DD;                            // 16 MB

    k_prep_M   <<<DD * DD / 256, 256, 0, stream>>>(out_w, in_proj_w, Mbuf);
    k_prep_P   <<<DD * DD / 256, 256, 0, stream>>>(fc1_w, Mbuf, Pbuf);
    k_prep_bias<<<1, 256, 0, stream>>>(out_w, out_b, in_proj_b, fc1_w, fc1_b, Pbuf,
                                       b_ctx, b_body, b_face, b_text, hb);
    k_prep_Wbig<<<DD * 17, 256, 0, stream>>>(Pbuf, w_ctx, w_body, w_face, w_text, wbig);
    k_main     <<<NBLK, 256, 0, stream>>>(x_ctx, x_body, x_face, x_text, wbig, hb,
                                          hpre, psum, psq);
    k_stats    <<<1, 256, 0, stream>>>(psum, psq, bn_g, bn_b, scale, shift);
    k_head     <<<NB / 8, 256, 0, stream>>>(hpre, scale, shift, fc2_w, fc2_b, outp);
}
